// NeighborhoodAwareVAE_11828339933907
// MI455X (gfx1250) — matrix operand order repaired and checked by reading, not compiled, not run on hardware
//
#include <hip/hip_runtime.h>
#include <stdint.h>

#define NN 100000
#define NE 800000

typedef __attribute__((ext_vector_type(16))) __bf16 v16bf;
typedef __attribute__((ext_vector_type(8)))  float  v8f;

union frag { v16bf v; uint32_t u[8]; };

__device__ __forceinline__ uint32_t pk2bf(float a, float b) {
  uint32_t ua = __float_as_uint(a), ub = __float_as_uint(b);
  ua = (ua + 0x7FFFu + ((ua >> 16) & 1u)) >> 16;
  ub = (ub + 0x7FFFu + ((ub >> 16) & 1u)) >> 16;
  return ua | (ub << 16);
}
__device__ __forceinline__ uint16_t f2bf(float a) {
  uint32_t u = __float_as_uint(a);
  return (uint16_t)((u + 0x7FFFu + ((u >> 16) & 1u)) >> 16);
}

// ---- prep: pack x [N][64] f32 -> [N][32] u32 (bf16 pairs) -------------------
__global__ void k_pack_x(const float* __restrict__ x, uint32_t* __restrict__ xp) {
  int id = blockIdx.x * blockDim.x + threadIdx.x;
  if (id >= NN * 32) return;
  int n = id >> 5, p = id & 31;
  const float* r = x + (size_t)n * 64 + 2 * p;
  xp[id] = pk2bf(r[0], r[1]);
}

// ---- prep: pack weight [K][Nout] f32 -> [Kpad/2][Nout] u32, zero-pad K ------
__global__ void k_pack_w(const float* __restrict__ W, uint32_t* __restrict__ dst,
                         int K, int Nout, int Kpad) {
  int id = blockIdx.x * blockDim.x + threadIdx.x;
  int total = (Kpad >> 1) * Nout;
  if (id >= total) return;
  int p = id / Nout, n = id - p * Nout;
  int k0 = 2 * p, k1 = k0 + 1;
  float a = (k0 < K) ? W[(size_t)k0 * Nout + n] : 0.f;
  float b = (k1 < K) ? W[(size_t)k1 * Nout + n] : 0.f;
  dst[id] = pk2bf(a, b);
}

// ---- segment mean: one wave per node (edge_dst sorted), lane owns 2 cols ----
__global__ void k_segavg(const float* __restrict__ x, const int* __restrict__ esrc,
                         const int* __restrict__ edst, uint32_t* __restrict__ navg) {
  int lane = threadIdx.x & 31;
  int wid  = (blockIdx.x * blockDim.x + threadIdx.x) >> 5;
  int nw   = (gridDim.x * blockDim.x) >> 5;
  for (int node = wid; node < NN; node += nw) {
    int lo = 0, hi = NE;
    while (lo < hi) { int mid = (lo + hi) >> 1; if (edst[mid] <  node) lo = mid + 1; else hi = mid; }
    int s0 = lo; hi = NE;
    while (lo < hi) { int mid = (lo + hi) >> 1; if (edst[mid] <= node) lo = mid + 1; else hi = mid; }
    int s1 = lo;
    float ax = 0.f, ay = 0.f;
    for (int e = s0; e < s1; e++) {
      const float2 v = ((const float2*)(x + (size_t)esrc[e] * 64))[lane];
      ax += v.x; ay += v.y;
    }
    float inv = 1.f / fmaxf((float)(s1 - s0), 1.f);
    navg[(size_t)node * 32 + lane] = pk2bf(ax * inv, ay * inv);
  }
}

// ---- WMMA fragment helpers --------------------------------------------------
// B (32x16 bf16): lane<16 -> N=lane, K pairs 0..3,8..11 ; lane>=16 -> K pairs 4..7,12..15
__device__ __forceinline__ v16bf ldB(const uint32_t* w, int Nout, int kc, int nt, int lane) {
  int n  = nt * 16 + (lane & 15);
  int pb = kc * 16 + ((lane & 16) ? 4 : 0);
  frag f;
#pragma unroll
  for (int j = 0; j < 8; j++) f.u[j] = w[(pb + j + (j & 4)) * Nout + n];
  return f.v;
}
// A (16x32 bf16) from packed global [row][32] u32
__device__ __forceinline__ v16bf ldA_g(const uint32_t* pk, int r0, int kc, int lane) {
  int m = lane & 15, h = (lane >> 4) & 1;
  const uint32_t* b = pk + (size_t)(r0 + m) * 32 + kc * 16;
  frag f;
#pragma unroll
  for (int j = 0; j < 8; j++) {
    int p = (j < 4) ? (4 * h + j) : (8 + 4 * h + (j - 4));
    f.u[j] = b[p];
  }
  return f.v;
}
// A (16x32 bf16) from LDS act tile, row stride 72 elems (conflict-free)
__device__ __forceinline__ v16bf ldA_s(const uint16_t* a, int colbase, int lane) {
  int m = lane & 15, h = (lane >> 4) & 1;
  const uint16_t* row = a + m * 72 + colbase;
  frag f;
#pragma unroll
  for (int j = 0; j < 8; j++) {
    int k = (j < 4) ? (8 * h + 2 * j) : (16 + 8 * h + 2 * (j - 4));
    f.u[j] = *(const uint32_t*)(row + k);
  }
  return f.v;
}
// C initialized with per-column bias
__device__ __forceinline__ v8f biasC(const float* sb, int off, int nt, int lane) {
  float b = sb[off + nt * 16 + (lane & 15)];
  v8f c;
#pragma unroll
  for (int i = 0; i < 8; i++) c[i] = b;
  return c;
}
// D tile (f32) -> act LDS (bf16): lane=col, VGPR v = row v+8h
__device__ __forceinline__ void stAct(uint16_t* a, int nt, int lane, v8f d) {
  int n = lane & 15, h = (lane >> 4) & 1;
#pragma unroll
  for (int v = 0; v < 8; v++) a[(v + 8 * h) * 72 + nt * 16 + n] = f2bf(d[v]);
}

#define WMMA(a, b, c) \
  __builtin_amdgcn_wmma_f32_16x16x32_bf16(false, (a), false, (b), (short)0, (c), false, false)

// ---- fused forward: 4 waves/block, one 16-node tile per wave iteration ------
__global__ void __launch_bounds__(128) k_fused(
    const uint32_t* __restrict__ xp,   const uint32_t* __restrict__ navg,
    const uint32_t* __restrict__ pAgg, const uint32_t* __restrict__ pFus,
    const uint32_t* __restrict__ pFc1, const uint32_t* __restrict__ pFc21,
    const uint32_t* __restrict__ pFc22,const uint32_t* __restrict__ pFc3,
    const uint32_t* __restrict__ pFc4,
    const float* __restrict__ bAgg, const float* __restrict__ bFus,
    const float* __restrict__ bF1,  const float* __restrict__ bF21,
    const float* __restrict__ bF22, const float* __restrict__ bF3,
    const float* __restrict__ bF4,
    const float* __restrict__ eps, float* __restrict__ out) {
  __shared__ uint32_t wAgg [32 * 64];
  __shared__ uint32_t wFus [64 * 64];
  __shared__ uint32_t wFc1 [32 * 64];
  __shared__ uint32_t wFc21[32 * 16];
  __shared__ uint32_t wFc22[32 * 16];
  __shared__ uint32_t wFc3 [16 * 64];
  __shared__ uint32_t wFc4 [32 * 64];
  __shared__ float    sb[352];
  __shared__ __align__(16) uint16_t act[4][16 * 72];

  const int tid = threadIdx.x;
  for (int i = tid; i < 2048; i += 128) wAgg[i]  = pAgg[i];
  for (int i = tid; i < 4096; i += 128) wFus[i]  = pFus[i];
  for (int i = tid; i < 2048; i += 128) wFc1[i]  = pFc1[i];
  for (int i = tid; i <  512; i += 128) wFc21[i] = pFc21[i];
  for (int i = tid; i <  512; i += 128) wFc22[i] = pFc22[i];
  for (int i = tid; i < 1024; i += 128) wFc3[i]  = pFc3[i];
  for (int i = tid; i < 2048; i += 128) wFc4[i]  = pFc4[i];
  for (int i = tid; i < 64; i += 128) {
    sb[i] = bAgg[i]; sb[64 + i] = bFus[i]; sb[128 + i] = bF1[i];
    sb[224 + i] = bF3[i]; sb[288 + i] = bF4[i];
  }
  if (tid < 16) { sb[192 + tid] = bF21[tid]; sb[208 + tid] = bF22[tid]; }
  __syncthreads();

  const int lane = tid & 31;
  const int h = (lane >> 4) & 1, n = lane & 15;
  uint16_t* my = &act[tid >> 5][0];
  const int wid = blockIdx.x * 4 + (tid >> 5);
  const int nwaves = gridDim.x * 4;
  float* muO = out + (size_t)NN * 64;
  float* lvO = out + (size_t)NN * 80;

  for (int tile = wid; tile < NN / 16; tile += nwaves) {
    const int r0 = tile * 16;
    // --- neighbor encoder: relu(navg @ W_agg + b_agg)
    v16bf a0 = ldA_g(navg, r0, 0, lane), a1 = ldA_g(navg, r0, 1, lane);
#pragma unroll
    for (int t = 0; t < 4; t++) {
      v8f c = biasC(sb, 0, t, lane);
      c = WMMA(a0, ldB(wAgg, 64, 0, t, lane), c);
      c = WMMA(a1, ldB(wAgg, 64, 1, t, lane), c);
#pragma unroll
      for (int i = 0; i < 8; i++) c[i] = fmaxf(c[i], 0.f);
      stAct(my, t, lane, c);
    }
    // --- fusion: relu([x | nenc] @ W_fus + b_fus)
    v16bf x0 = ldA_g(xp, r0, 0, lane), x1 = ldA_g(xp, r0, 1, lane);
    v16bf e0 = ldA_s(my, 0, lane),     e1 = ldA_s(my, 32, lane);
    v8f xf[4];
#pragma unroll
    for (int t = 0; t < 4; t++) {
      v8f c = biasC(sb, 64, t, lane);
      c = WMMA(x0, ldB(wFus, 64, 0, t, lane), c);
      c = WMMA(x1, ldB(wFus, 64, 1, t, lane), c);
      c = WMMA(e0, ldB(wFus, 64, 2, t, lane), c);
      c = WMMA(e1, ldB(wFus, 64, 3, t, lane), c);
#pragma unroll
      for (int i = 0; i < 8; i++) c[i] = fmaxf(c[i], 0.f);
      xf[t] = c;
    }
#pragma unroll
    for (int t = 0; t < 4; t++) stAct(my, t, lane, xf[t]);
    // --- fc1
    v16bf f0 = ldA_s(my, 0, lane), f1 = ldA_s(my, 32, lane);
    v8f h1[4];
#pragma unroll
    for (int t = 0; t < 4; t++) {
      v8f c = biasC(sb, 128, t, lane);
      c = WMMA(f0, ldB(wFc1, 64, 0, t, lane), c);
      c = WMMA(f1, ldB(wFc1, 64, 1, t, lane), c);
#pragma unroll
      for (int i = 0; i < 8; i++) c[i] = fmaxf(c[i], 0.f);
      h1[t] = c;
    }
#pragma unroll
    for (int t = 0; t < 4; t++) stAct(my, t, lane, h1[t]);
    // --- fc21 (mu) / fc22 (logvar), Nout = 16
    v16bf g0 = ldA_s(my, 0, lane), g1 = ldA_s(my, 32, lane);
    v8f mu = biasC(sb, 192, 0, lane);
    mu = WMMA(g0, ldB(wFc21, 16, 0, 0, lane), mu);
    mu = WMMA(g1, ldB(wFc21, 16, 1, 0, lane), mu);
    v8f lv = biasC(sb, 208, 0, lane);
    lv = WMMA(g0, ldB(wFc22, 16, 0, 0, lane), lv);
    lv = WMMA(g1, ldB(wFc22, 16, 1, 0, lane), lv);
    // --- reparameterize, emit mu/logvar
    v8f z;
#pragma unroll
    for (int v = 0; v < 8; v++) {
      int row = r0 + v + 8 * h;
      muO[(size_t)row * 16 + n] = mu[v];
      lvO[(size_t)row * 16 + n] = lv[v];
      float ep = eps[(size_t)row * 16 + n];
      z[v] = mu[v] + ep * __expf(0.5f * lv[v]);
    }
    stAct(my, 0, lane, z);                         // z -> cols 0..15
#pragma unroll
    for (int v = 0; v < 8; v++) my[(v + 8 * h) * 72 + 16 + n] = (uint16_t)0;  // K pad
    v16bf az = ldA_s(my, 0, lane);
    // --- fc3 (K padded 16->32)
#pragma unroll
    for (int t = 0; t < 4; t++) {
      v8f c = biasC(sb, 224, t, lane);
      c = WMMA(az, ldB(wFc3, 64, 0, t, lane), c);
#pragma unroll
      for (int i = 0; i < 8; i++) c[i] = fmaxf(c[i], 0.f);
      stAct(my, t, lane, c);
    }
    // --- fc4 + sigmoid -> recon
    v16bf q0 = ldA_s(my, 0, lane), q1 = ldA_s(my, 32, lane);
#pragma unroll
    for (int t = 0; t < 4; t++) {
      v8f c = biasC(sb, 288, t, lane);
      c = WMMA(q0, ldB(wFc4, 64, 0, t, lane), c);
      c = WMMA(q1, ldB(wFc4, 64, 1, t, lane), c);
#pragma unroll
      for (int v = 0; v < 8; v++) {
        int row = r0 + v + 8 * h;
        out[(size_t)row * 64 + t * 16 + n] = 1.f / (1.f + __expf(-c[v]));
      }
    }
  }
}

extern "C" void kernel_launch(void* const* d_in, const int* in_sizes, int n_in,
                              void* d_out, int out_size, void* d_ws, size_t ws_size,
                              hipStream_t stream) {
  (void)in_sizes; (void)n_in; (void)out_size; (void)ws_size;
  const float* x    = (const float*)d_in[0];
  const int*   esrc = (const int*)d_in[1];
  const int*   edst = (const int*)d_in[2];
  const float* eps  = (const float*)d_in[3];
  const float* W_agg = (const float*)d_in[4];  const float* b_agg = (const float*)d_in[5];
  const float* W_fus = (const float*)d_in[6];  const float* b_fus = (const float*)d_in[7];
  const float* W1    = (const float*)d_in[8];  const float* b1    = (const float*)d_in[9];
  const float* W21   = (const float*)d_in[10]; const float* b21   = (const float*)d_in[11];
  const float* W22   = (const float*)d_in[12]; const float* b22   = (const float*)d_in[13];
  const float* W3    = (const float*)d_in[14]; const float* b3    = (const float*)d_in[15];
  const float* W4    = (const float*)d_in[16]; const float* b4    = (const float*)d_in[17];
  float* out = (float*)d_out;

  uint32_t* xp    = (uint32_t*)d_ws;
  uint32_t* navg  = xp    + (size_t)NN * 32;
  uint32_t* pAgg  = navg  + (size_t)NN * 32;
  uint32_t* pFus  = pAgg  + 32 * 64;
  uint32_t* pFc1  = pFus  + 64 * 64;
  uint32_t* pFc21 = pFc1  + 32 * 64;
  uint32_t* pFc22 = pFc21 + 32 * 16;
  uint32_t* pFc3  = pFc22 + 32 * 16;
  uint32_t* pFc4  = pFc3  + 16 * 64;

  k_pack_x<<<(NN * 32 + 255) / 256, 256, 0, stream>>>(x, xp);
  k_pack_w<<<8,  256, 0, stream>>>(W_agg, pAgg, 64, 64, 64);
  k_pack_w<<<16, 256, 0, stream>>>(W_fus, pFus, 128, 64, 128);
  k_pack_w<<<8,  256, 0, stream>>>(W1,  pFc1,  64, 64, 64);
  k_pack_w<<<2,  256, 0, stream>>>(W21, pFc21, 64, 16, 64);
  k_pack_w<<<2,  256, 0, stream>>>(W22, pFc22, 64, 16, 64);
  k_pack_w<<<4,  256, 0, stream>>>(W3,  pFc3,  16, 64, 32);
  k_pack_w<<<8,  256, 0, stream>>>(W4,  pFc4,  64, 64, 64);
  k_segavg<<<1024, 256, 0, stream>>>(x, esrc, edst, navg);
  k_fused<<<512, 128, 0, stream>>>(xp, navg, pAgg, pFus, pFc1, pFc21, pFc22, pFc3, pFc4,
                                   b_agg, b_fus, b1, b21, b22, b3, b4, eps, out);
}
